// AllYouNeedAttention_16870631539192
// MI455X (gfx1250) — compile-verified, matrix operand order checked by reading, not run on hardware
//
#include <hip/hip_runtime.h>
#include <hip/hip_bf16.h>

// ---------------------------------------------------------------------------
// Multi-head attention forward for MI455X (gfx1250, wave32, WMMA).
//   out = [ h (B,S,D) | softmax(scores) (B,H,S,S) | scores (B,H,S,S) ]
// Pipeline:
//   k0a: x (fp32) -> xb (bf16)                  [one-time convert, 24MB traffic]
//   k0b: wq/wk/wv (fp32, KxN) -> w^T (bf16 NxK) [so B-fragments load contiguously]
//   k1 : QKV projection, bf16 WMMA. A tile double-buffered into LDS with
//        GLOBAL_LOAD_ASYNC_TO_LDS_B128 + S_WAIT_ASYNCCNT (fallback: sync copy).
//        Q/K -> (B,H,S,W) bf16, V -> (B,H,W,S) bf16 (transposed).
//   k2 : fused QK^T + mask + softmax + PV per 32-query stripe; the 32x2048 fp32
//        score stripe lives entirely in LDS (256KB < 320KB/WGP), so scores are
//        written once and never re-read from HBM. Streaming outputs use NT
//        stores to keep L2 for the 24MB Q/K/V^T working set.
// ---------------------------------------------------------------------------

typedef __attribute__((ext_vector_type(16))) __bf16       v16bf;
typedef __attribute__((ext_vector_type(8)))  float        v8f;
typedef __attribute__((ext_vector_type(8)))  unsigned int v8u;
typedef __attribute__((ext_vector_type(4)))  unsigned int u32x4;
typedef __attribute__((ext_vector_type(4)))  int          v4i;

constexpr int Bb = 2, Ss = 2048, Dd = 1024, Hh = 16, Ww = 64;

#if defined(__gfx1250__) &&                                                    \
    __has_builtin(__builtin_amdgcn_global_load_async_to_lds_b128) &&           \
    __has_builtin(__builtin_amdgcn_s_wait_asynccnt)
#define USE_ASYNC_LDS 1
#endif

__device__ __forceinline__ unsigned int pack2_bf16(float a, float b) {
  unsigned ia = __builtin_bit_cast(unsigned, a);
  unsigned ib = __builtin_bit_cast(unsigned, b);
  ia = (ia + 0x7FFFu + ((ia >> 16) & 1u)) >> 16;   // round-to-nearest-even
  ib = (ib + 0x7FFFu + ((ib >> 16) & 1u)) >> 16;
  return ia | (ib << 16);
}
__device__ __forceinline__ unsigned short bf16_1(float a) {
  unsigned ia = __builtin_bit_cast(unsigned, a);
  return (unsigned short)((ia + 0x7FFFu + ((ia >> 16) & 1u)) >> 16);
}

// A/B fragment from a row-major bf16 matrix (ISA 7.12.2 16-bit A layout):
// lane L: row = row0 + (L%16); VGPRs 0-3 hold k = kb + (L/16)*8 + 0..7,
// VGPRs 4-7 hold k = kb + 16 + (L/16)*8 + 0..7.  Two 16B loads per lane.
__device__ __forceinline__ v16bf frag_b16(const unsigned short* base, int ld,
                                          int row0, int kb) {
  const int lane = threadIdx.x & 31, m = lane & 15, half = lane >> 4;
  const unsigned short* rp = base + (size_t)(row0 + m) * ld + kb + half * 8;
  u32x4 lo = *(const u32x4*)(rp);
  u32x4 hi = *(const u32x4*)(rp + 16);
  v8u raw = {lo.x, lo.y, lo.z, lo.w, hi.x, hi.y, hi.z, hi.w};
  return __builtin_bit_cast(v16bf, raw);
}

// Same fragment layout, sourced from fp32 (LDS score stripe) with on-the-fly
// bf16 conversion (only used once per PV k-step).
__device__ __forceinline__ v16bf frag_f32(const float* base, int ld,
                                          int row0, int kb) {
  const int lane = threadIdx.x & 31, m = lane & 15, half = lane >> 4;
  const float* rp = base + (size_t)(row0 + m) * ld;
  v8u raw;
#pragma unroll
  for (int g = 0; g < 2; ++g) {
    const float* p = rp + kb + g * 16 + half * 8;
    float4 f0 = *(const float4*)(p);
    float4 f1 = *(const float4*)(p + 4);
    raw[g * 4 + 0] = pack2_bf16(f0.x, f0.y);
    raw[g * 4 + 1] = pack2_bf16(f0.z, f0.w);
    raw[g * 4 + 2] = pack2_bf16(f1.x, f1.y);
    raw[g * 4 + 3] = pack2_bf16(f1.z, f1.w);
  }
  return __builtin_bit_cast(v16bf, raw);
}

__device__ __forceinline__ v8f wmma_bf16(v16bf a, v16bf b, v8f c) {
  // (neg_a, A, neg_b, B, c_mod, C, reuse_a, reuse_b)
  return __builtin_amdgcn_wmma_f32_16x16x32_bf16(false, a, false, b, (short)0, c,
                                                 false, false);
}

__device__ __forceinline__ float wave_max(float v) {
#pragma unroll
  for (int o = 16; o > 0; o >>= 1) v = fmaxf(v, __shfl_xor(v, o, 32));
  return v;
}
__device__ __forceinline__ float wave_sum(float v) {
#pragma unroll
  for (int o = 16; o > 0; o >>= 1) v += __shfl_xor(v, o, 32);
  return v;
}

// 16B tile-staging copy global->LDS: async DMA path on CDNA5 (ASYNCcnt),
// synchronous fallback otherwise.  Builtin signature (from diagnostics):
//   (int4 AS1* gsrc, int4 AS3* lds, imm offset, imm cpol)
__device__ __forceinline__ void stage16(unsigned short* lds_dst,
                                        const unsigned short* gsrc) {
#if defined(USE_ASYNC_LDS)
  __builtin_amdgcn_global_load_async_to_lds_b128(
      (__attribute__((address_space(1))) v4i*)(v4i*)gsrc,
      (__attribute__((address_space(3))) v4i*)(v4i*)lds_dst, 0, 0);
#else
  *(u32x4*)lds_dst = *(const u32x4*)gsrc;
#endif
}
template <int N>
__device__ __forceinline__ void stage_wait() {
#if defined(USE_ASYNC_LDS)
  __builtin_amdgcn_s_wait_asynccnt(N);   // imm operand
#endif
}

// ---------------------------------------------------------------------------
// k0a: x fp32 -> bf16 (same layout).  8 elements / thread.
// ---------------------------------------------------------------------------
__global__ __launch_bounds__(256) void cvt_x_kernel(
    const float* __restrict__ x, unsigned short* __restrict__ xb) {
  const size_t i = ((size_t)blockIdx.x * 256 + threadIdx.x) * 8;
  float4 f0 = *(const float4*)(x + i);
  float4 f1 = *(const float4*)(x + i + 4);
  u32x4 o = {pack2_bf16(f0.x, f0.y), pack2_bf16(f0.z, f0.w),
             pack2_bf16(f1.x, f1.y), pack2_bf16(f1.z, f1.w)};
  *(u32x4*)(xb + i) = o;
}

// ---------------------------------------------------------------------------
// k0b: w (fp32, K x N) -> w^T (bf16, N x K), 32x32 tiles through LDS.
// grid = (K/32, N/32, 3), block = 256 (32x8 logical).
// ---------------------------------------------------------------------------
__global__ __launch_bounds__(256) void cvt_wT_kernel(
    const float* __restrict__ wq, const float* __restrict__ wk,
    const float* __restrict__ wv, unsigned short* __restrict__ wqt,
    unsigned short* __restrict__ wkt, unsigned short* __restrict__ wvt) {
  __shared__ float tile[32][33];
  const int z = blockIdx.z;
  const float* w = (z == 0) ? wq : (z == 1) ? wk : wv;
  unsigned short* o = (z == 0) ? wqt : (z == 1) ? wkt : wvt;
  const int k0 = blockIdx.x * 32, n0 = blockIdx.y * 32;
  const int tx = threadIdx.x & 31, ty = threadIdx.x >> 5;
#pragma unroll
  for (int i = 0; i < 4; ++i) {
    const int k = ty * 4 + i;
    tile[k][tx] = w[(size_t)(k0 + k) * Dd + n0 + tx];   // coalesced along n
  }
  __syncthreads();
#pragma unroll
  for (int i = 0; i < 4; ++i) {
    const int n = ty * 4 + i;
    o[(size_t)(n0 + n) * Dd + k0 + tx] = bf16_1(tile[tx][n]);  // coalesced along k
  }
}

// ---------------------------------------------------------------------------
// k1: QKV projection.  grid = (D/64, B*S/64, 3), block = 256 (8 waves).
// 64x64 tile of xb @ w[z] + b[z].  A tile (64x32 bf16, 4KB) double-buffered
// into LDS via async-to-LDS; B fragments load straight from global w^T.
// ---------------------------------------------------------------------------
__global__ __launch_bounds__(256) void qkv_proj_kernel(
    const unsigned short* __restrict__ xb,
    const unsigned short* __restrict__ wqt, const float* __restrict__ bq,
    const unsigned short* __restrict__ wkt, const float* __restrict__ bk,
    const unsigned short* __restrict__ wvt, const float* __restrict__ bvv,
    unsigned short* __restrict__ qo, unsigned short* __restrict__ ko,
    unsigned short* __restrict__ vto) {
  __shared__ unsigned short xs[2][64][40];   // padded ld=40 (80B) keeps 16B align

  const int z = blockIdx.z;
  const unsigned short* wT = (z == 0) ? wqt : (z == 1) ? wkt : wvt;
  const float* bias = (z == 0) ? bq : (z == 1) ? bk : bvv;
  unsigned short* outp = (z == 0) ? qo : (z == 1) ? ko : vto;

  const int n0 = blockIdx.x * 64;
  const int m0 = blockIdx.y * 64;
  const int tid = threadIdx.x;
  const int lane = tid & 31;
  const int wid = tid >> 5;
  const int nt = wid & 3;
  const int mt0 = (wid >> 2) * 2;

  // staging role: one 16B chunk (8 bf16) per thread per k-tile
  const int srow = tid >> 2, sc0 = (tid & 3) * 8;
  const unsigned short* xsrc = xb + (size_t)(m0 + srow) * Dd + sc0;

  v8f acc[2] = {{0, 0, 0, 0, 0, 0, 0, 0}, {0, 0, 0, 0, 0, 0, 0, 0}};

  stage16(&xs[0][srow][sc0], xsrc);   // prime buffer 0 (kb = 0)

  for (int kb = 0; kb < Dd; kb += 32) {
    const int cur = (kb >> 5) & 1, nxt = cur ^ 1;
    if (kb + 32 < Dd) {
      stage16(&xs[nxt][srow][sc0], xsrc + kb + 32);  // prefetch next tile
      stage_wait<1>();                               // cur tile has landed
    } else {
      stage_wait<0>();
    }
    __syncthreads();
    v16bf Bf = frag_b16(wT, Dd, n0 + nt * 16, kb);   // global, contiguous 16B
#pragma unroll
    for (int p = 0; p < 2; ++p) {
      v16bf Af = frag_b16(&xs[cur][0][0], 40, (mt0 + p) * 16, 0);
      acc[p] = wmma_bf16(Af, Bf, acc[p]);
    }
    __syncthreads();   // done reading cur before it gets re-staged
  }

  // Epilogue: + bias, convert to bf16, scatter into head-major layouts.
  const int half = lane >> 4, nloc = lane & 15;
  const int n = n0 + nt * 16 + nloc;
  const float bval = bias[n];
  const int head = n >> 6, w = n & 63;
#pragma unroll
  for (int p = 0; p < 2; ++p) {
#pragma unroll
    for (int i = 0; i < 8; ++i) {
      const int r = m0 + (mt0 + p) * 16 + half * 8 + i;  // global row in (B*S)
      const int b = r >> 11, s = r & (Ss - 1);
      const unsigned short hv = bf16_1(acc[p][i] + bval);
      size_t idx;
      if (z < 2)  // Q, K: (B,H,S,W)
        idx = (((size_t)(b * Hh + head)) * Ss + s) * Ww + w;
      else        // V transposed: (B,H,W,S) so PV B-fragments load contiguously
        idx = (((size_t)(b * Hh + head)) * Ww + w) * Ss + s;
      outp[idx] = hv;
    }
  }
}

// ---------------------------------------------------------------------------
// k2: fused attention.  grid = (S/32, B*H), block = 256 (8 waves).
// ---------------------------------------------------------------------------
__global__ __launch_bounds__(256) void attn_kernel(
    const unsigned short* __restrict__ qm, const unsigned short* __restrict__ km,
    const unsigned short* __restrict__ vtm, const float* __restrict__ mask,
    float* __restrict__ hout, float* __restrict__ soft_out,
    float* __restrict__ score_out) {
  extern __shared__ float Stile[];  // 32 x 2048 fp32 (256KB)

  const int bh = blockIdx.y;
  const int b = bh >> 4, hIdx = bh & 15;
  const int q0 = blockIdx.x * 32;
  const int tid = threadIdx.x, lane = tid & 31, wid = tid >> 5;
  const int half = lane >> 4, nloc = lane & 15;

  const unsigned short* qb = qm + (size_t)bh * Ss * Ww;
  const unsigned short* kbm = km + (size_t)bh * Ss * Ww;
  const unsigned short* vb = vtm + (size_t)bh * Ww * Ss;

  // Q A-fragments for this stripe: 2 M-tiles x 2 k-steps (W=64).
  v16bf Aq[2][2];
#pragma unroll
  for (int mt = 0; mt < 2; ++mt)
#pragma unroll
    for (int ks = 0; ks < 2; ++ks)
      Aq[mt][ks] = frag_b16(qb, Ww, q0 + mt * 16, ks * 32);

  // ---- phase 1: scores = Q K^T / 8 - 10000*(1-mask) -------------------------
  for (int j = 0; j < 16; ++j) {
    const int ncol0 = wid * 256 + j * 16;
    v8f a0 = {0, 0, 0, 0, 0, 0, 0, 0}, a1 = {0, 0, 0, 0, 0, 0, 0, 0};
#pragma unroll
    for (int ks = 0; ks < 2; ++ks) {
      v16bf Bf = frag_b16(kbm, Ww, ncol0, ks * 32);  // K rows = B^T columns
      a0 = wmma_bf16(Aq[0][ks], Bf, a0);
      a1 = wmma_bf16(Aq[1][ks], Bf, a1);
    }
    const int col = ncol0 + nloc;
    const float madd = -10000.0f * (1.0f - mask[b * Ss + col]);
#pragma unroll
    for (int i = 0; i < 8; ++i) {
      const int m0r = half * 8 + i;
      const float s0 = a0[i] * 0.125f + madd;
      Stile[m0r * Ss + col] = s0;
      __builtin_nontemporal_store(
          s0, &score_out[((size_t)bh * Ss + q0 + m0r) * Ss + col]);
      const int m1r = 16 + half * 8 + i;
      const float s1 = a1[i] * 0.125f + madd;
      Stile[m1r * Ss + col] = s1;
      __builtin_nontemporal_store(
          s1, &score_out[((size_t)bh * Ss + q0 + m1r) * Ss + col]);
    }
  }
  __syncthreads();

  // ---- phase 2: softmax over each of the 32 rows (4 rows per wave) ----------
#pragma unroll
  for (int rr = 0; rr < 4; ++rr) {
    const int r = wid * 4 + rr;
    float* row = Stile + (size_t)r * Ss;
    float mx = -3.4e38f;
    for (int t = 0; t < Ss / 32; ++t) mx = fmaxf(mx, row[lane + t * 32]);
    mx = wave_max(mx);
    float sum = 0.0f;
    for (int t = 0; t < Ss / 32; ++t) {
      const int i2 = lane + t * 32;
      const float e = __expf(row[i2] - mx);
      row[i2] = e;
      sum += e;
    }
    sum = wave_sum(sum);
    const float inv = 1.0f / sum;
    float* srow = soft_out + ((size_t)bh * Ss + q0 + r) * Ss;
    for (int t = 0; t < Ss / 32; ++t) {
      const int i2 = lane + t * 32;
      const float p = row[i2] * inv;
      row[i2] = p;                                // P stays in LDS for phase 3
      __builtin_nontemporal_store(p, &srow[i2]);  // streamed out once
    }
  }
  __syncthreads();

  // ---- phase 3: h = P @ V  (M=32, N=64, K=2048) -----------------------------
  const int mt = wid >> 2, nt = wid & 3;
  v8f acc = {0, 0, 0, 0, 0, 0, 0, 0};
  for (int kb = 0; kb < Ss; kb += 32) {
    v16bf Ap = frag_f32(Stile, Ss, mt * 16, kb);
    v16bf Bv = frag_b16(vb, Ss, nt * 16, kb);
    acc = wmma_bf16(Ap, Bv, acc);
  }
  const int d = hIdx * Ww + nt * 16 + nloc;
#pragma unroll
  for (int i = 0; i < 8; ++i) {
    const int qrow = q0 + mt * 16 + half * 8 + i;
    __builtin_nontemporal_store(acc[i],
                                &hout[((size_t)b * Ss + qrow) * Dd + d]);
  }
}

// ---------------------------------------------------------------------------
extern "C" void kernel_launch(void* const* d_in, const int* in_sizes, int n_in,
                              void* d_out, int out_size, void* d_ws,
                              size_t ws_size, hipStream_t stream) {
  const float* x    = (const float*)d_in[0];
  const float* mask = (const float*)d_in[1];
  const float* wq   = (const float*)d_in[2];
  const float* bq   = (const float*)d_in[3];
  const float* wk   = (const float*)d_in[4];
  const float* bk   = (const float*)d_in[5];
  const float* wv   = (const float*)d_in[6];
  const float* bv   = (const float*)d_in[7];

  float* out = (float*)d_out;
  const size_t hElems = (size_t)Bb * Ss * Dd;       // 4M
  const size_t sElems = (size_t)Bb * Hh * Ss * Ss;  // 134M
  float* hout = out;
  float* soft = out + hElems;
  float* scor = out + hElems + sElems;

  // Workspace layout (bf16): Q,K (B,H,S,W), V^T (B,H,W,S), xb, wq^T, wk^T, wv^T
  const size_t qkvN = (size_t)Bb * Hh * Ss * Ww;  // 4M
  unsigned short* qws  = (unsigned short*)d_ws;
  unsigned short* kws  = qws + qkvN;
  unsigned short* vtws = kws + qkvN;
  unsigned short* xbws = vtws + qkvN;                      // 4M
  unsigned short* wqtw = xbws + (size_t)Bb * Ss * Dd;      // 1M each
  unsigned short* wktw = wqtw + (size_t)Dd * Dd;
  unsigned short* wvtw = wktw + (size_t)Dd * Dd;

  cvt_x_kernel<<<dim3(((size_t)Bb * Ss * Dd) / (256 * 8)), 256, 0, stream>>>(
      x, xbws);
  cvt_wT_kernel<<<dim3(Dd / 32, Dd / 32, 3), 256, 0, stream>>>(
      wq, wk, wv, wqtw, wktw, wvtw);
  qkv_proj_kernel<<<dim3(Dd / 64, (Bb * Ss) / 64, 3), 256, 0, stream>>>(
      xbws, wqtw, bq, wktw, bk, wvtw, bv, qws, kws, vtws);

  const size_t shBytes = (size_t)32 * Ss * sizeof(float);  // 256 KB < 320 KB/WGP
  (void)hipFuncSetAttribute((const void*)attn_kernel,
                            hipFuncAttributeMaxDynamicSharedMemorySize,
                            (int)shBytes);
  attn_kernel<<<dim3(Ss / 32, Bb * Hh), 256, shBytes, stream>>>(
      qws, kws, vtws, mask, hout, soft, scor);
}
